// CausalSelfAttention_91139206021480
// MI455X (gfx1250) — compile-verified, matrix operand order checked by reading, not run on hardware
//
#include <hip/hip_runtime.h>
#include <math.h>

// ---------------------------------------------------------------------------
// CDNA5 (gfx1250) causal self-attention, bf16 WMMA + f32 accumulate.
// ---------------------------------------------------------------------------

typedef __bf16 bf16;
typedef __attribute__((ext_vector_type(16))) __bf16       v16bf;
typedef __attribute__((ext_vector_type(8)))  float        v8f;
typedef __attribute__((ext_vector_type(4)))  unsigned int v4u;

#define NEGBIG (-1.0e30f)

__device__ __forceinline__ bf16 f2bf(float f) {
  unsigned u = __builtin_bit_cast(unsigned, f);
  u += 0x7fffu + ((u >> 16) & 1u);              // round-to-nearest-even
  unsigned short h = (unsigned short)(u >> 16);
  return __builtin_bit_cast(bf16, h);
}

// --- 16x32 bf16 A/B fragment load (ISA 7.12.2 layout) ----------------------
// A (and B given Bt is stored N x K row-major): lane l in 0..15 holds row l,
// K = {hs*8 .. hs*8+7} and {16+hs*8 ..}, hs = lane/16.
__device__ __forceinline__ v16bf load_frag16(const bf16* base, int ld) {
  const int lane = threadIdx.x & 31;
  const int r  = lane & 15;
  const int hs = lane >> 4;
  const bf16* p = base + (size_t)r * ld + hs * 8;
  union { v4u u[2]; v16bf v; } x;
  x.u[0] = *(const v4u*)(p);        // K = hs*8 + 0..7
  x.u[1] = *(const v4u*)(p + 16);   // K = 16 + hs*8 + 0..7
  return x.v;
}

__device__ __forceinline__ v8f wmma_bf16(v16bf a, v16bf b, v8f c) {
  return __builtin_amdgcn_wmma_f32_16x16x32_bf16(
      /*neg_a=*/false, a, /*neg_b=*/false, b,
      /*c_mod=*/(short)0, c, /*reuse_a=*/false, /*reuse_b=*/false);
}

__device__ __forceinline__ void store_val(float* p, float v) { *p = v; }
__device__ __forceinline__ void store_val(bf16* p, float v) { *p = f2bf(v); }

// ---------------------------------------------------------------------------
// fp32 -> bf16 conversion (grid-stride)
// ---------------------------------------------------------------------------
__global__ void cvt_f32_to_bf16(const float* __restrict__ in,
                                bf16* __restrict__ out, int n) {
  int i = blockIdx.x * blockDim.x + threadIdx.x;
  int stride = gridDim.x * blockDim.x;
  for (; i < n; i += stride) out[i] = f2bf(in[i]);
}

// ---------------------------------------------------------------------------
// Generic WMMA GEMM: C[M,N] = A[M,K] * Bt[N,K]^T ; f32 accumulate.
// One wave owns a 16(M) x 64(N) tile of C; data streamed from L2.
// ---------------------------------------------------------------------------
template <typename OutT>
__global__ void gemm_bf16_kernel(const bf16* __restrict__ A,
                                 const bf16* __restrict__ Bt,
                                 OutT* __restrict__ C,
                                 int M, int N, int K) {
  const int wave = blockIdx.x * (blockDim.x >> 5) + (threadIdx.x >> 5);
  const int tilesN = N / 64;
  const int tm = wave / tilesN;
  const int tn = wave % tilesN;
  if (tm * 16 >= M) return;

  v8f acc[4] = {};
  const bf16* arow = A + (size_t)tm * 16 * K;
  for (int k = 0; k < K; k += 32) {
    v16bf a = load_frag16(arow + k, K);
#pragma unroll
    for (int j = 0; j < 4; ++j) {
      v16bf b = load_frag16(Bt + (size_t)(tn * 64 + j * 16) * K + k, K);
      acc[j] = wmma_bf16(a, b, acc[j]);
    }
  }
  const int lane = threadIdx.x & 31;
  const int n = lane & 15, mh = lane >> 4;
#pragma unroll
  for (int j = 0; j < 4; ++j) {
    OutT* d = C + (size_t)tm * 16 * N + tn * 64 + j * 16;
#pragma unroll
    for (int g = 0; g < 8; ++g)
      store_val(&d[(size_t)(mh * 8 + g) * N + n], acc[j][g]);
  }
}

// ---------------------------------------------------------------------------
// V transpose: qkv[b*T+t][1024+c] -> vT[b][c][t]   (32x32 LDS tiles)
// ---------------------------------------------------------------------------
__global__ void transpose_v_kernel(const bf16* __restrict__ qkv,
                                   bf16* __restrict__ vT) {
  __shared__ bf16 tile[32][33];
  const int b = blockIdx.z;
  const int t0 = blockIdx.x * 32, c0 = blockIdx.y * 32;
#pragma unroll
  for (int i = threadIdx.y; i < 32; i += 8)
    tile[i][threadIdx.x] =
        qkv[(size_t)(b * 4096 + t0 + i) * 1536 + 1024 + c0 + threadIdx.x];
  __syncthreads();
#pragma unroll
  for (int i = threadIdx.y; i < 32; i += 8)
    vT[(size_t)b * 512 * 4096 + (size_t)(c0 + i) * 4096 + t0 + threadIdx.x] =
        tile[threadIdx.x][i];
}

// ---------------------------------------------------------------------------
// Flash-style causal attention.
// Grid: (T/16, B). Block: 128 threads = 4 waves.
// A workgroup owns one 16-query-row block; wave w owns channels [w*128, +128).
// Q (16x512) is staged in LDS once; each K tile (32x512) is staged in LDS
// cooperatively per key step (removes the 4x redundant L2 traffic and keeps
// fragment loads on the DS path so nothing spills to scratch).
// ---------------------------------------------------------------------------
__global__ void attn_kernel(const bf16* __restrict__ qkv,
                            const bf16* __restrict__ vT,
                            bf16* __restrict__ attn_out, float scale) {
  const int T = 4096, LDQKV = 1536;
  const int qb = blockIdx.x;
  const int b = blockIdx.y;
  const int wave = threadIdx.x >> 5;
  const int lane = threadIdx.x & 31;
  const int n = lane & 15, mh = lane >> 4;
  const int tid = threadIdx.x;

  __shared__ alignas(16) bf16 q_lds[16 * 512];   // 16 KB
  __shared__ alignas(16) bf16 k_lds[32 * 512];   // 32 KB
  __shared__ alignas(16) bf16 p_lds[4][520];     // per-wave P staging (16x32)
  bf16* pbuf = &p_lds[wave][0];

  // ---- stage Q block once: rows qb*16..+15, cols 0..511 -------------------
  {
    const bf16* qsrc = qkv + (size_t)(b * T + qb * 16) * LDQKV;
    v4u* dst = (v4u*)q_lds;
    for (int i = tid; i < 16 * 64; i += 128) {
      int r = i >> 6, c = i & 63;
      dst[i] = ((const v4u*)(qsrc + (size_t)r * LDQKV))[c];
    }
  }
  __syncthreads();

  const bf16* vtb = vT + (size_t)b * 512 * T;    // [C][T]

  v8f acc[8] = {};
  float mrow[8], lrow[8];
#pragma unroll
  for (int g = 0; g < 8; ++g) { mrow[g] = NEGBIG; lrow[g] = 0.0f; }

  const int kmax = qb * 16 + 15;                 // causal limit
  for (int kb = 0; kb * 32 <= kmax; ++kb) {
    // ---- cooperative stage of K tile: rows kb*32..+31, cols 512..1023 -----
    {
      const bf16* ksrc = qkv + (size_t)(b * T + kb * 32) * LDQKV + 512;
      v4u* dst = (v4u*)k_lds;
      for (int i = tid; i < 32 * 64; i += 128) {
        int r = i >> 6, c = i & 63;
        dst[i] = ((const v4u*)(ksrc + (size_t)r * LDQKV))[c];
      }
    }
    __syncthreads();

    // ---- scores S[16 x 32] = Q(16x512) * K(32x512)^T : two 16x16 tiles ----
    v8f s0 = {}, s1 = {};
    for (int kc = 0; kc < 512; kc += 32) {
      v16bf aq = load_frag16(q_lds + kc, 512);
      v16bf b0 = load_frag16(k_lds + kc, 512);
      v16bf b1 = load_frag16(k_lds + 16 * 512 + kc, 512);
      s0 = wmma_bf16(aq, b0, s0);
      s1 = wmma_bf16(aq, b1, s1);
    }
    // ---- mask + online softmax (row = mh*8+g, col = n in C-layout) --------
    float p0[8], p1[8], fsc[8];
#pragma unroll
    for (int g = 0; g < 8; ++g) {
      const int qrow = qb * 16 + mh * 8 + g;
      float x0 = s0[g] * scale;
      float x1 = s1[g] * scale;
      if (kb * 32 + n > qrow) x0 = NEGBIG;
      if (kb * 32 + 16 + n > qrow) x1 = NEGBIG;
      float mx = fmaxf(x0, x1);
#pragma unroll
      for (int msk = 1; msk <= 8; msk <<= 1)
        mx = fmaxf(mx, __shfl_xor(mx, msk, 32));
      const float mnew = fmaxf(mrow[g], mx);
      const float fg = __expf(mrow[g] - mnew);
      p0[g] = __expf(x0 - mnew);
      p1[g] = __expf(x1 - mnew);
      float rs = p0[g] + p1[g];
#pragma unroll
      for (int msk = 1; msk <= 8; msk <<= 1) rs += __shfl_xor(rs, msk, 32);
      lrow[g] = lrow[g] * fg + rs;
      mrow[g] = mnew;
      fsc[g] = fg;
    }
    // ---- rescale output accumulators --------------------------------------
#pragma unroll
    for (int t = 0; t < 8; ++t)
#pragma unroll
      for (int g = 0; g < 8; ++g) acc[t][g] *= fsc[g];
    // ---- P: C-layout -> row-major LDS -> A-fragment -----------------------
#pragma unroll
    for (int g = 0; g < 8; ++g) {
      pbuf[(mh * 8 + g) * 32 + n] = f2bf(p0[g]);
      pbuf[(mh * 8 + g) * 32 + 16 + n] = f2bf(p1[g]);
    }
    v16bf pf = load_frag16(pbuf, 32);  // same wave: DS dependency handled
    // ---- O += P(16x32) * V(32x128-chunk) ----------------------------------
#pragma unroll
    for (int t = 0; t < 8; ++t) {
      const int ch = wave * 128 + t * 16;
      v16bf vf = load_frag16(vtb + (size_t)ch * T + kb * 32, T);
      acc[t] = wmma_bf16(pf, vf, acc[t]);
    }
    __syncthreads();   // protect k_lds before next iteration overwrites it
  }
  // ---- normalize + store bf16 ---------------------------------------------
  float inv[8];
#pragma unroll
  for (int g = 0; g < 8; ++g) inv[g] = 1.0f / lrow[g];
  bf16* orow = attn_out + (size_t)(b * T + qb * 16) * 512 + wave * 128;
#pragma unroll
  for (int t = 0; t < 8; ++t)
#pragma unroll
    for (int g = 0; g < 8; ++g)
      orow[(size_t)(mh * 8 + g) * 512 + t * 16 + n] = f2bf(acc[t][g] * inv[g]);
}

// ---------------------------------------------------------------------------
extern "C" void kernel_launch(void* const* d_in, const int* in_sizes, int n_in,
                              void* d_out, int out_size, void* d_ws, size_t ws_size,
                              hipStream_t stream) {
  (void)in_sizes; (void)n_in; (void)out_size; (void)ws_size;
  const float* x     = (const float*)d_in[0];  // [4,4096,512]
  const float* wqkv  = (const float*)d_in[1];  // [1536,512]
  const float* wproj = (const float*)d_in[2];  // [512,512]
  float* out = (float*)d_out;                  // [4,4096,512] fp32

  char* ws = (char*)d_ws;
  size_t off = 0;
  bf16* x_bf     = (bf16*)(ws + off); off += (size_t)16384 * 512 * 2;   // 16 MB
  bf16* wqkv_bf  = (bf16*)(ws + off); off += (size_t)1536 * 512 * 2;
  bf16* wproj_bf = (bf16*)(ws + off); off += (size_t)512 * 512 * 2;
  bf16* qkv_bf   = (bf16*)(ws + off); off += (size_t)16384 * 1536 * 2;  // 48 MB
  bf16* vT       = (bf16*)(ws + off); off += (size_t)4 * 512 * 4096 * 2;
  bf16* attn_bf  = (bf16*)(ws + off); off += (size_t)16384 * 512 * 2;

  // 1) fp32 -> bf16
  cvt_f32_to_bf16<<<1024, 256, 0, stream>>>(x, x_bf, 16384 * 512);
  cvt_f32_to_bf16<<<256, 256, 0, stream>>>(wqkv, wqkv_bf, 1536 * 512);
  cvt_f32_to_bf16<<<128, 256, 0, stream>>>(wproj, wproj_bf, 512 * 512);

  // 2) QKV projection: [16384,512] x [1536,512]^T -> [16384,1536] bf16
  {
    int waves = (16384 / 16) * (1536 / 64);
    gemm_bf16_kernel<bf16><<<(waves + 7) / 8, 256, 0, stream>>>(
        x_bf, wqkv_bf, qkv_bf, 16384, 1536, 512);
  }

  // 3) V transpose -> vT[b][c][t]
  transpose_v_kernel<<<dim3(4096 / 32, 512 / 32, 4), dim3(32, 8), 0, stream>>>(
      qkv_bf, vT);

  // 4) causal attention (scale = 1/sqrt(512))
  attn_kernel<<<dim3(4096 / 16, 4), 128, 0, stream>>>(
      qkv_bf, vT, attn_bf, 0.044194173824159216f);

  // 5) output projection: [16384,512] x [512,512]^T -> fp32 d_out
  {
    int waves = (16384 / 16) * (512 / 64);
    gemm_bf16_kernel<float><<<(waves + 7) / 8, 256, 0, stream>>>(
        attn_bf, wproj_bf, out, 16384, 512, 512);
  }
}
